// RNet_55456617726106
// MI455X (gfx1250) — compile-verified, matrix operand order checked by reading, not run on hardware
//
#include <hip/hip_runtime.h>
#include <hip/hip_bf16.h>
#include <math.h>

// ---------------------------------------------------------------------------
// R-Net forward for MI455X (gfx1250, wave32, WMMA).
// All matmuls are y = x * W^T with W stored [N,K] row-major -> both WMMA
// operands load contiguous 16B strips per lane. f16 operands, f32 accum.
// Partial tiles use row-index CLAMPING instead of guarded loads: out-of-range
// output rows/cols compute garbage that is never stored, so every K-loop is
// branch-free (no EXEC juggling inside the hot loop).
// ---------------------------------------------------------------------------

typedef __attribute__((ext_vector_type(16))) _Float16 v16h;
typedef __attribute__((ext_vector_type(8)))  _Float16 v8h;
typedef __attribute__((ext_vector_type(8)))  float    v8f;

#define EMB   300
#define HID   768
#define DD    1536          // 2*H
#define GH_N  2304          // 3*H
#define QQ    48
#define PP    128
#define BB    8

// ---- WMMA micro-tile helpers ----------------------------------------------

// Load one lane's 16-half strip of a 16x32 (MxK) operand. Row-major source.
// K layout per CDNA5 ISA: lanes 0-15 hold K k0..k0+7 and k0+16..k0+23;
// lanes 16-31 hold k0+8..k0+15 and k0+24..k0+31.
__device__ __forceinline__ v16h load_strip16(const _Float16* rowp, int k0, int hs) {
    v8h lo = *(const v8h*)(rowp + k0 + hs * 8);
    v8h hi = *(const v8h*)(rowp + k0 + 16 + hs * 8);
    v16h r;
#pragma unroll
    for (int i = 0; i < 8; ++i) { r[i] = lo[i]; r[i + 8] = hi[i]; }
    return r;
}

// Dual N-tile accumulate: acc0 += A*W[n0..n0+16)^T, acc1 += A*W[n0+16..n0+32)^T.
// Both N-tiles must be fully in range. A rows are clamped to Mrows-1: lanes
// beyond M read duplicate rows, producing garbage only in output rows that
// store_tile never writes. Branch-free K-loop; EXEC all-ones at the WMMAs.
__device__ __forceinline__ void wmma_tile2(const _Float16* A, int lda, int Mrows, int m0,
                                           const _Float16* W, int ldw, int n0,
                                           int Kp, v8f& acc0, v8f& acc1) {
    const int lane = threadIdx.x & 31;
    const int hs   = lane >> 4;
    const int r    = lane & 15;
    const int am   = min(m0 + r, Mrows - 1);
    const _Float16* ap = A + (size_t)am * lda;
    const _Float16* w0 = W + (size_t)(n0 + r) * ldw;
    const _Float16* w1 = w0 + (size_t)16 * ldw;
    for (int k0 = 0; k0 < Kp; k0 += 32) {
        v16h a  = load_strip16(ap, k0, hs);
        v16h b0 = load_strip16(w0, k0, hs);
        v16h b1 = load_strip16(w1, k0, hs);
        __builtin_prefetch(w0 + k0 + 512, 0, 1);   // global_prefetch_b8
        __builtin_prefetch(w1 + k0 + 512, 0, 1);
        acc0 = __builtin_amdgcn_wmma_f32_16x16x32_f16(false, a, false, b0,
                                                      (short)0, acc0, false, false);
        acc1 = __builtin_amdgcn_wmma_f32_16x16x32_f16(false, a, false, b1,
                                                      (short)0, acc1, false, false);
    }
}

// Single tile with clamped M and N row indices (ragged N edge).
__device__ __forceinline__ v8f wmma_tile_acc(const _Float16* A, int lda, int Mrows, int m0,
                                             const _Float16* W, int ldw, int Nrows, int n0,
                                             int Kp, v8f acc) {
    const int lane = threadIdx.x & 31;
    const int hs   = lane >> 4;
    const int r    = lane & 15;
    const int am   = min(m0 + r, Mrows - 1);
    const int wn   = min(n0 + r, Nrows - 1);
    const _Float16* ap = A + (size_t)am * lda;
    const _Float16* wp = W + (size_t)wn * ldw;
    for (int k0 = 0; k0 < Kp; k0 += 32) {
        v16h a = load_strip16(ap, k0, hs);
        v16h b = load_strip16(wp, k0, hs);
        acc = __builtin_amdgcn_wmma_f32_16x16x32_f16(false, a, false, b,
                                                     (short)0, acc, false, false);
    }
    return acc;
}

// C/D 16x16 f32 layout: lanes 0-15 hold N=lane, M=j; lanes 16-31 N=lane-16, M=8+j.
__device__ __forceinline__ void store_tile(float* C, int ldc, int Mrows, int Nrows,
                                           int m0, int n0, v8f acc,
                                           const float* bias, int act) {
    const int lane  = threadIdx.x & 31;
    const int n     = n0 + (lane & 15);
    const int mbase = m0 + ((lane >> 4) ? 8 : 0);
    if (n >= Nrows) return;
    const float bv = bias ? bias[n] : 0.f;
#pragma unroll
    for (int j = 0; j < 8; ++j) {
        const int m = mbase + j;
        if (m < Mrows) {
            float v = acc[j] + bv;
            if (act == 1) v = fmaxf(v, 0.f);
            else if (act == 2) v = tanhf(v);
            C[(size_t)m * ldc + n] = v;
        }
    }
}

__device__ __forceinline__ float sigmoidf_(float x) { return 1.f / (1.f + __expf(-x)); }

// ---- Generic wide GEMM: C[M,N] = act(A[M,Kp] * W[N,Kp]^T + bias) ----------
// One wave owns a 16x32 output slab (two 16x16 N-tiles sharing the A strips).

__global__ __launch_bounds__(256)
void gemm_f16w(const _Float16* __restrict__ A, int lda,
               const _Float16* __restrict__ W, int ldw,
               const float* __restrict__ bias,
               float* __restrict__ C, int ldc,
               int M, int N, int Kp, int act) {
    const int wave    = (blockIdx.x * blockDim.x + threadIdx.x) >> 5;
    const int tilesN  = (N + 15) >> 4;
    const int tilesM  = (M + 15) >> 4;
    const int pairsN  = (tilesN + 1) >> 1;
    if (wave >= tilesM * pairsN) return;          // wave-uniform exit
    const int tm = wave / pairsN;
    const int n0 = (wave % pairsN) * 32;
    const int m0 = tm * 16;

    if (n0 + 32 <= N) {                           // both N-tiles interior
        v8f a0 = {}, a1 = {};
        wmma_tile2(A, lda, M, m0, W, ldw, n0, Kp, a0, a1);
        store_tile(C, ldc, M, N, m0, n0,      a0, bias, act);
        store_tile(C, ldc, M, N, m0, n0 + 16, a1, bias, act);
    } else {                                      // ragged N edge
        v8f acc = {};
        acc = wmma_tile_acc(A, lda, M, m0, W, ldw, N, n0, Kp, acc);
        store_tile(C, ldc, M, N, m0, n0, acc, bias, act);
        if (n0 + 16 < N) {
            v8f acc1 = {};
            acc1 = wmma_tile_acc(A, lda, M, m0, W, ldw, N, n0 + 16, Kp, acc1);
            store_tile(C, ldc, M, N, m0, n0 + 16, acc1, bias, act);
        }
    }
}

// ---- f32 -> f16 with row padding (rows get stride ldp, zero-filled) -------

__global__ void cvt_pad_kernel(const float* __restrict__ src, _Float16* __restrict__ dst,
                               int rows, int K, int ldp) {
    size_t idx = (size_t)blockIdx.x * blockDim.x + threadIdx.x;
    size_t total = (size_t)rows * ldp;
    if (idx >= total) return;
    int r = (int)(idx / ldp), k = (int)(idx % ldp);
    dst[idx] = (k < K) ? (_Float16)src[(size_t)r * K + k] : (_Float16)0.f;
}

// ---- One bi-GRU layer: blockIdx.x = direction, whole T-step recurrence ----
// gx (precomputed x*Wih^T + bih) in [T][B][3H]; h kept in LDS (f32 + f16).

__global__ __launch_bounds__(512)
void gru_layer_kernel(const float* __restrict__ gx_f, const float* __restrict__ gx_b,
                      const _Float16* __restrict__ Whh_f, const _Float16* __restrict__ Whh_b,
                      const float* __restrict__ bhh_f, const float* __restrict__ bhh_b,
                      const float* __restrict__ h0,     // [2][B][H] (fwd,bwd for this layer)
                      float* __restrict__ out,          // [T][B][2H]
                      int T, float* __restrict__ ghws)  // [2][B][3H] scratch
{
    const int dir = blockIdx.x;
    const float*    gx  = dir ? gx_b  : gx_f;
    const _Float16* Whh = dir ? Whh_b : Whh_f;
    const float*    bhh = dir ? bhh_b : bhh_f;
    const float*    h0d = h0 + (size_t)dir * BB * HID;
    float*          gh  = ghws + (size_t)dir * BB * GH_N;
    const int colBase   = dir ? HID : 0;

    __shared__ _Float16 h16[BB][HID];
    __shared__ float    h32[BB][HID];

    for (int i = threadIdx.x; i < BB * HID; i += blockDim.x) {
        float v = h0d[i];
        h32[i / HID][i % HID] = v;
        h16[i / HID][i % HID] = (_Float16)v;
    }
    __syncthreads();

    const int wave = threadIdx.x >> 5, nwaves = blockDim.x >> 5;
    const int pairsN = (GH_N / 16) / 2;   // 72

    for (int t = 0; t < T; ++t) {
        const int ti = dir ? (T - 1 - t) : t;

        // Stage A: gh = h * Whh^T   (M=8, N=2304, K=768), LDS A, 2 tiles/wave
        for (int p = wave; p < pairsN; p += nwaves) {
            v8f a0 = {}, a1 = {};
            wmma_tile2(&h16[0][0], HID, BB, 0, Whh, HID, p * 32, HID, a0, a1);
            store_tile(gh, GH_N, BB, GH_N, 0, p * 32,      a0, nullptr, 0);
            store_tile(gh, GH_N, BB, GH_N, 0, p * 32 + 16, a1, nullptr, 0);
        }
        __threadfence_block();
        __syncthreads();

        // Stage B: GRU pointwise update (PyTorch gate order r,z,n)
        const float* gxt = gx + (size_t)ti * BB * GH_N;
        for (int i = threadIdx.x; i < BB * HID; i += blockDim.x) {
            const int b = i / HID, j = i % HID;
            const float* gxb = gxt + (size_t)b * GH_N;
            const float* ghb = gh  + (size_t)b * GH_N;
            float r = sigmoidf_(gxb[j]           + ghb[j]           + bhh[j]);
            float z = sigmoidf_(gxb[HID + j]     + ghb[HID + j]     + bhh[HID + j]);
            float n = tanhf   (gxb[2 * HID + j] + r * (ghb[2 * HID + j] + bhh[2 * HID + j]));
            float hnew = (1.f - z) * n + z * h32[b][j];
            h32[b][j] = hnew;
            h16[b][j] = (_Float16)hnew;
            out[(size_t)ti * BB * DD + (size_t)b * DD + colBase + j] = hnew;
        }
        __threadfence_block();
        __syncthreads();
    }
}

// ---- PQ matcher: single block, 128 sequential steps -----------------------

__global__ __launch_bounds__(512)
void matcher_kernel(const float* __restrict__ po, const float* __restrict__ qo,
                    const float* __restrict__ Qw, const float* __restrict__ v0,
                    const _Float16* __restrict__ Wpu, const _Float16* __restrict__ Wpv,
                    const _Float16* __restrict__ Wihm, const _Float16* __restrict__ Whhm,
                    const float* __restrict__ bih, const float* __restrict__ bhh,
                    float* __restrict__ vw, float* __restrict__ avw,
                    float* __restrict__ cw, float* __restrict__ gatesw, int T)
{
    __shared__ _Float16 ut16[BB][DD];
    __shared__ _Float16 v16s[BB][DD];
    __shared__ float sL[QQ][BB];
    __shared__ float aL[QQ][BB];
    __shared__ float dA[BB];
    __shared__ float ssum[BB];

    const int tid = threadIdx.x;
    const int wave = tid >> 5, lane = tid & 31, nwaves = blockDim.x >> 5;

    for (int i = tid; i < BB * DD; i += blockDim.x) {
        float v = v0[i];
        vw[i] = v;
        v16s[i / DD][i % DD] = (_Float16)v;
    }
    __threadfence_block();
    __syncthreads();

    for (int t = 0; t < T; ++t) {
        // ut = po[t] -> LDS f16
        const float* ut = po + (size_t)t * BB * DD;
        for (int i = tid; i < BB * DD; i += blockDim.x)
            ut16[i / DD][i % DD] = (_Float16)ut[i];
        __syncthreads();

        // av = ut*Wpu^T + v*Wpv^T   (M=8, N=1536, K=1536 twice), 2 tiles/wave
        for (int p = wave; p < (DD / 16) / 2; p += nwaves) {
            v8f a0 = {}, a1 = {};
            wmma_tile2(&ut16[0][0], DD, BB, 0, Wpu, DD, p * 32, DD, a0, a1);
            wmma_tile2(&v16s[0][0], DD, BB, 0, Wpv, DD, p * 32, DD, a0, a1);
            store_tile(avw, DD, BB, DD, 0, p * 32,      a0, nullptr, 0);
            store_tile(avw, DD, BB, DD, 0, p * 32 + 16, a1, nullptr, 0);
        }
        __threadfence_block();
        __syncthreads();

        // s[q][b] = Qw[q][b]·v[b]  (384 dots) ; dA[b] = av[b]·v[b] (8 dots)
        for (int item = wave; item < QQ * BB + BB; item += nwaves) {
            float part = 0.f;
            if (item < QQ * BB) {
                const int q = item >> 3, b = item & 7;
                const float* qp = Qw + ((size_t)q * BB + b) * DD;
                const float* vb = vw + (size_t)b * DD;
                for (int d = lane; d < DD; d += 32) part += qp[d] * vb[d];
                for (int off = 16; off; off >>= 1) part += __shfl_xor(part, off, 32);
                if (lane == 0) sL[q][b] = part;
            } else {
                const int b = item - QQ * BB;
                const float* ab = avw + (size_t)b * DD;
                const float* vb = vw  + (size_t)b * DD;
                for (int d = lane; d < DD; d += 32) part += ab[d] * vb[d];
                for (int off = 16; off; off >>= 1) part += __shfl_xor(part, off, 32);
                if (lane == 0) dA[b] = part;
            }
        }
        __syncthreads();

        if (tid < BB) {
            float s = 0.f;
            for (int q = 0; q < QQ; ++q) s += sL[q][tid] + dA[tid];
            ssum[tid] = s;
        }
        __syncthreads();
        for (int i = tid; i < QQ * BB; i += blockDim.x) {
            const int q = i >> 3, b = i & 7;
            aL[q][b] = (sL[q][b] + dA[b]) / ssum[b];
        }
        __syncthreads();

        // c[b][d] = sum_q a[q][b] * qo[q][b][d]
        for (int i = tid; i < BB * DD; i += blockDim.x) {
            const int b = i / DD, d = i % DD;
            float acc = 0.f;
            for (int q = 0; q < QQ; ++q)
                acc += aL[q][b] * qo[((size_t)q * BB + b) * DD + d];
            cw[i] = acc;
        }
        __threadfence_block();
        __syncthreads();

        // gates = v*Wih^T + ut*Whh^T   (M=8, N=6144, K=1536 twice), 2 tiles/wave
        for (int p = wave; p < (4 * DD / 16) / 2; p += nwaves) {
            v8f a0 = {}, a1 = {};
            wmma_tile2(&v16s[0][0], DD, BB, 0, Wihm, DD, p * 32, DD, a0, a1);
            wmma_tile2(&ut16[0][0], DD, BB, 0, Whhm, DD, p * 32, DD, a0, a1);
            store_tile(gatesw, 4 * DD, BB, 4 * DD, 0, p * 32,      a0, nullptr, 0);
            store_tile(gatesw, 4 * DD, BB, 4 * DD, 0, p * 32 + 16, a1, nullptr, 0);
        }
        __threadfence_block();
        __syncthreads();

        // LSTM-cell-ish update: i,f,g,o gate order
        for (int i = tid; i < BB * DD; i += blockDim.x) {
            const int b = i / DD, j = i % DD;
            const float* gb = gatesw + (size_t)b * 4 * DD;
            float ig = gb[j]          + bih[j]          + bhh[j];
            float fg = gb[DD + j]     + bih[DD + j]     + bhh[DD + j];
            float gg = gb[2 * DD + j] + bih[2 * DD + j] + bhh[2 * DD + j];
            float og = gb[3 * DD + j] + bih[3 * DD + j] + bhh[3 * DD + j];
            float cn = sigmoidf_(fg) * cw[i] + sigmoidf_(ig) * tanhf(gg);
            float vn = sigmoidf_(og) * tanhf(cn);
            vw[i] = vn;
            v16s[b][j] = (_Float16)vn;
        }
        __threadfence_block();
        __syncthreads();
    }
}

// ---------------------------------------------------------------------------
// Host side
// ---------------------------------------------------------------------------

// Input flattening (setup_inputs dict insertion order, params recursively):
// 0:p 1:q 2:h0p 3:h0q 4:v0
// 5..28  : penc  l0f(Wih,Whh,bih,bhh) l0b l1f l1b l2f l2b
// 29..52 : qenc  same
// 53..59 : match Wqu Wpu Wpv Wih Whh bih bhh
// 60..67 : fc    W1 b1 W2 b2 W3 b3 W4 b4
enum { IN_P = 0, IN_Q = 1, IN_H0P = 2, IN_H0Q = 3, IN_V0 = 4,
       IN_PENC = 5, IN_QENC = 29, IN_MATCH = 53, IN_FC = 60 };

static inline int kpad32(int k) { return (k + 31) & ~31; }

extern "C" void kernel_launch(void* const* d_in, const int* in_sizes, int n_in,
                              void* d_out, int out_size, void* d_ws, size_t ws_size,
                              hipStream_t stream) {
    (void)in_sizes; (void)n_in; (void)out_size; (void)ws_size;
    auto F = [&](int i) { return (const float*)d_in[i]; };

    char* wsb = (char*)d_ws;
    size_t off = 0;
    auto alloc = [&](size_t bytes) -> void* {
        off = (off + 255) & ~(size_t)255;
        void* p = wsb + off;
        off += bytes;
        return p;
    };
    auto allocH = [&](size_t elems) { return (_Float16*)alloc(elems * sizeof(_Float16)); };
    auto allocF = [&](size_t elems) { return (float*)alloc(elems * sizeof(float)); };

    auto cvt = [&](const float* src, _Float16* dst, int rows, int K, int ldp) {
        size_t total = (size_t)rows * ldp;
        int blocks = (int)((total + 255) / 256);
        cvt_pad_kernel<<<blocks, 256, 0, stream>>>(src, dst, rows, K, ldp);
    };
    auto gemm = [&](const _Float16* A, int lda, const _Float16* W, int ldw,
                    const float* bias, float* C, int ldc, int M, int N, int Kp, int act) {
        int pairs = ((M + 15) / 16) * ((((N + 15) / 16) + 1) / 2);
        int blocks = (pairs + 7) / 8;           // 8 waves / 256 threads per block
        gemm_f16w<<<blocks, 256, 0, stream>>>(A, lda, W, ldw, bias, C, ldc, M, N, Kp, act);
    };

    // Shared activation scratch
    _Float16* xin16 = allocH((size_t)PP * BB * DD);     // layer input, f16 (max size)
    float*    gx_f  = allocF((size_t)PP * BB * GH_N);
    float*    gx_b  = allocF((size_t)PP * BB * GH_N);
    float*    ghws  = allocF((size_t)2 * BB * GH_N);
    float*    po    = allocF((size_t)PP * BB * DD);
    float*    qo    = allocF((size_t)QQ * BB * DD);

    // -------- Encoders --------
    for (int e = 0; e < 2; ++e) {
        const int T    = e ? QQ : PP;
        const int rows = T * BB;
        const int base = e ? IN_QENC : IN_PENC;
        const float* x   = F(e ? IN_Q : IN_P);
        const float* h0  = F(e ? IN_H0Q : IN_H0P);
        float*       out = e ? qo : po;

        cvt(x, xin16, rows, EMB, kpad32(EMB));
        for (int l = 0; l < 3; ++l) {
            const int I  = (l == 0) ? EMB : DD;
            const int Ip = kpad32(I);
            const int bf = base + (l * 2 + 0) * 4;    // fwd: Wih,Whh,bih,bhh
            const int bb = base + (l * 2 + 1) * 4;    // bwd

            _Float16* Wih_f16 = allocH((size_t)GH_N * Ip);
            _Float16* Wih_b16 = allocH((size_t)GH_N * Ip);
            _Float16* Whh_f16 = allocH((size_t)GH_N * HID);
            _Float16* Whh_b16 = allocH((size_t)GH_N * HID);

            cvt(F(bf + 0), Wih_f16, GH_N, I, Ip);
            cvt(F(bb + 0), Wih_b16, GH_N, I, Ip);
            gemm(xin16, Ip, Wih_f16, Ip, F(bf + 2), gx_f, GH_N, rows, GH_N, Ip, 0);
            gemm(xin16, Ip, Wih_b16, Ip, F(bb + 2), gx_b, GH_N, rows, GH_N, Ip, 0);

            cvt(F(bf + 1), Whh_f16, GH_N, HID, HID);
            cvt(F(bb + 1), Whh_b16, GH_N, HID, HID);
            gru_layer_kernel<<<2, 512, 0, stream>>>(gx_f, gx_b, Whh_f16, Whh_b16,
                                                    F(bf + 3), F(bb + 3),
                                                    h0 + (size_t)(2 * l) * BB * HID,
                                                    out, T, ghws);
            if (l < 2) cvt(out, xin16, rows, DD, DD);
        }
    }

    // -------- Matcher --------
    _Float16* qo16  = allocH((size_t)QQ * BB * DD);
    _Float16* Wqu16 = allocH((size_t)DD * DD);
    _Float16* Wpu16 = allocH((size_t)DD * DD);
    _Float16* Wpv16 = allocH((size_t)DD * DD);
    _Float16* Wihm16 = allocH((size_t)4 * DD * DD);
    _Float16* Whhm16 = allocH((size_t)4 * DD * DD);
    float* Qw     = allocF((size_t)QQ * BB * DD);
    float* vw     = allocF((size_t)BB * DD);
    float* avw    = allocF((size_t)BB * DD);
    float* cw     = allocF((size_t)BB * DD);
    float* gatesw = allocF((size_t)BB * 4 * DD);

    cvt(qo, qo16, QQ * BB, DD, DD);
    cvt(F(IN_MATCH + 0), Wqu16, DD, DD, DD);
    gemm(qo16, DD, Wqu16, DD, nullptr, Qw, DD, QQ * BB, DD, DD, 0);
    cvt(F(IN_MATCH + 1), Wpu16, DD, DD, DD);
    cvt(F(IN_MATCH + 2), Wpv16, DD, DD, DD);
    cvt(F(IN_MATCH + 3), Wihm16, 4 * DD, DD, DD);
    cvt(F(IN_MATCH + 4), Whhm16, 4 * DD, DD, DD);
    matcher_kernel<<<1, 512, 0, stream>>>(po, qo, Qw, F(IN_V0),
                                          Wpu16, Wpv16, Wihm16, Whhm16,
                                          F(IN_MATCH + 5), F(IN_MATCH + 6),
                                          vw, avw, cw, gatesw, PP);

    // -------- FC stack: relu, tanh, tanh, tanh --------
    const int fcN[4] = {2000, 1500, 1000, 600};
    const int fcK[4] = {DD, 2000, 1500, 1000};
    float* fcin = vw;
    for (int i = 0; i < 4; ++i) {
        const int Kp = kpad32(fcK[i]);
        _Float16* A16 = allocH((size_t)BB * Kp);
        _Float16* W16 = allocH((size_t)fcN[i] * Kp);
        cvt(fcin, A16, BB, fcK[i], Kp);
        cvt(F(IN_FC + 2 * i), W16, fcN[i], fcK[i], Kp);
        float* Cout = (i == 3) ? (float*)d_out : allocF((size_t)BB * fcN[i]);
        gemm(A16, Kp, W16, Kp, F(IN_FC + 2 * i + 1), Cout, fcN[i],
             BB, fcN[i], Kp, (i == 0) ? 1 : 2);
        fcin = Cout;
    }
}
